// FullDiscretizerWLast_34763465294234
// MI455X (gfx1250) — matrix operand order repaired and checked by reading, not compiled, not run on hardware
//
#include <hip/hip_runtime.h>

#define B_SZ   4096
#define NI     1024
#define HID    2048
#define T_STEPS 20
#define DSTEP  48
#define NACT   45
#define NPAD   48

typedef __attribute__((ext_vector_type(16))) __bf16        v16bf;
typedef __attribute__((ext_vector_type(8)))  float         v8f;
typedef __attribute__((ext_vector_type(4)))  unsigned int  u32x4;

union Frag { v16bf bf; u32x4 u[2]; };

__device__ __forceinline__ unsigned short f2bf(float f) {
  unsigned int u = __builtin_bit_cast(unsigned int, f);
  u += 0x7FFFu + ((u >> 16) & 1u);   // round-to-nearest-even
  return (unsigned short)(u >> 16);
}

// ---------------- conversion kernels ----------------
__global__ void cvt_fc_kernel(const float* __restrict__ src, unsigned short* __restrict__ dst) {
  int i = blockIdx.x * 256 + threadIdx.x;
  if (i < B_SZ * NI) dst[i] = f2bf(src[i]);
}

// W1T[n][k] = W1[k][n], n<2048, k<1024 (bf16, row-major over K for B-fragment loads)
__global__ void cvt_w1t_kernel(const float* __restrict__ W1, unsigned short* __restrict__ W1T) {
  int i = blockIdx.x * 256 + threadIdx.x;
  if (i < HID * NI) {
    int n = i / NI, k = i % NI;
    W1T[i] = f2bf(W1[(size_t)k * HID + n]);
  }
}

// W2T[n][k] = W2[k][n] for n<45, zero-pad n=45..47
__global__ void cvt_w2t_kernel(const float* __restrict__ W2, unsigned short* __restrict__ W2T) {
  int i = blockIdx.x * 256 + threadIdx.x;
  if (i < NPAD * HID) {
    int n = i / HID, k = i % HID;
    W2T[i] = (n < NACT) ? f2bf(W2[(size_t)k * NACT + n]) : (unsigned short)0;
  }
}

// ---------------- recurrence (scan) ----------------
__global__ void scan_kernel(const float* __restrict__ camera,
                            const int* __restrict__ fb, const int* __restrict__ lr,
                            const int* __restrict__ jp, const int* __restrict__ ss,
                            const int* __restrict__ atk,
                            float* __restrict__ step48, int* __restrict__ dacBuf,
                            int* __restrict__ aliveBuf) {
  int b = blockIdx.x * blockDim.x + threadIdx.x;
  if (b >= B_SZ) return;
  const float cam0 = camera[2 * b], cam1 = camera[2 * b + 1];
  int acts[5] = { fb[b], lr[b], jp[b], ss[b], atk[b] };
  const int lo[5] = {6, 8, 10, 11, 14};
  const int oh[5] = {2, 5, 8, 10, 14};    // one-hot base offset in step record
  int rdac[5] = {0, 0, 0, 0, 0};
  float ac0 = 0.f, ac1 = 0.f;
  float delta = 0.0625f;                  // MIN_DELTA = 0.5 / 2^3
  bool chosen = false, committed = false;
  int cam_steps = 0;

  for (int t = 0; t < T_STEPS; ++t) {
    const bool alive = !committed;
    float* sp = step48 + ((size_t)t * B_SZ + b) * DSTEP;
    sp[0] = ac0; sp[1] = ac1;
    #pragma unroll
    for (int i = 2; i < 46; ++i) sp[i] = 0.f;
    #pragma unroll
    for (int i = 0; i < 5; ++i) sp[oh[i] + rdac[i]] = 1.f;
    sp[46] = delta; sp[47] = chosen ? 1.f : 0.f;

    bool commit = ((fabsf(cam0) < 1e-5f && fabsf(cam1) < 1e-5f) || (cam_steps >= 6));
    #pragma unroll
    for (int i = 0; i < 5; ++i) commit = commit && (rdac[i] == acts[i]);

    int dac = 0;
    bool modified = commit;
    #pragma unroll
    for (int i = 0; i < 5; ++i) {
      const int a = acts[i], r = rdac[i];
      const bool disc = !modified && (r != a);
      const int val = (a == 0) ? (r - 1 + lo[i]) : (a - 1 + lo[i]);
      if (disc) { dac = val; rdac[i] = a; }
      modified = modified || disc;
    }
    // NOTE: reference compares both axes against ac0 (replicated exactly)
    const bool dmask = !modified && !chosen &&
        ((fabsf(cam0 - ac0) > delta * 2.f) || (fabsf(cam1 - ac0) > delta * 2.f));
    if (dmask) { dac = 5; delta = fminf(delta * 2.f, 0.5f); }
    modified = modified || dmask;
    const bool m = !modified;
    if (m) {
      const bool incX = (cam0 >= ac0), incY = (cam1 >= ac1);
      dac = 1 + (incX ? 1 : 0) + 2 * (incY ? 1 : 0);
      ac0 += incX ? delta : -delta;
      ac1 += incY ? delta : -delta;
      delta *= 0.5f;
      chosen = true;
      cam_steps += 1;
    }
    committed = committed || commit;
    dacBuf[(size_t)t * B_SZ + b] = dac;
    aliveBuf[(size_t)t * B_SZ + b] = alive ? 1 : 0;
  }
}

// ---------------- GEMM1 base: Hbase = fc(bf16) @ W1[:1024](bf16), fp32 acc ----------------
__global__ void __launch_bounds__(256) gemm_base_kernel(
    const unsigned short* __restrict__ fcB,
    const unsigned short* __restrict__ W1T,
    float* __restrict__ Hbase) {
  const int tid  = threadIdx.x;
  const int lane = tid & 31, wave = tid >> 5;
  const int hl   = lane >> 4, l16 = lane & 15;
  const int wm   = wave & 1,  wn  = wave >> 1;
  const int m0 = blockIdx.x * 128 + wm * 64;   // 4 M-tiles per wave
  const int n0 = blockIdx.y * 128 + wn * 32;   // 2 N-tiles per wave

  v8f zero = {};
  v8f acc[4][2];
  #pragma unroll
  for (int i = 0; i < 4; ++i)
    #pragma unroll
    for (int j = 0; j < 2; ++j) acc[i][j] = zero;

  for (int k0 = 0; k0 < NI; k0 += 32) {
    Frag a[4], bw[2];
    #pragma unroll
    for (int i = 0; i < 4; ++i) {
      // A 16x32 bf16 layout: lanes0-15 K=k0..k0+7 & +16..+23; lanes16-31 +8..+15 & +24..+31
      const unsigned short* p = fcB + (size_t)(m0 + i * 16 + l16) * NI + k0 + hl * 8;
      a[i].u[0] = *(const u32x4*)p;
      a[i].u[1] = *(const u32x4*)(p + 16);
    }
    #pragma unroll
    for (int j = 0; j < 2; ++j) {
      // B 32x16 bf16 layout: lanes0-15 K=k0..k0+7 & +16..+23 (N=lane); lanes16-31 +8..+15 & +24..+31
      const unsigned short* p = W1T + (size_t)(n0 + j * 16 + l16) * NI + k0 + hl * 8;
      bw[j].u[0] = *(const u32x4*)p;
      bw[j].u[1] = *(const u32x4*)(p + 16);
    }
    if (k0 + 32 < NI) {
      // locality 3 -> nearest scope: prefetch pulls into all cache levels (WGP$ + GL2)
      __builtin_prefetch(fcB + (size_t)(m0 + l16) * NI + k0 + 32, 0, 3);
      __builtin_prefetch(W1T + (size_t)(n0 + l16) * NI + k0 + 32, 0, 3);
    }
    #pragma unroll
    for (int i = 0; i < 4; ++i)
      #pragma unroll
      for (int j = 0; j < 2; ++j)
        acc[i][j] = __builtin_amdgcn_wmma_f32_16x16x32_bf16(
            false, a[i].bf, false, bw[j].bf, (short)0, acc[i][j], false, false);
  }

  // C/D layout: VGPR r -> row r (lanes 0-15) / row r+8 (lanes 16-31), col = lane&15
  #pragma unroll
  for (int i = 0; i < 4; ++i)
    #pragma unroll
    for (int j = 0; j < 2; ++j) {
      const int col = n0 + j * 16 + l16;
      #pragma unroll
      for (int r = 0; r < 8; ++r) {
        const int row = m0 + i * 16 + r + hl * 8;
        Hbase[(size_t)row * HID + col] = acc[i][j][r];
      }
    }
}

// ---------------- per-step fused: rank-48 update + swish + GEMM2 + CE ----------------
__global__ void __launch_bounds__(256) step_mlp_kernel(
    const float* __restrict__ Hbase, const float* __restrict__ step48,
    const int* __restrict__ dacBuf,
    const float* __restrict__ W1s,    // = W1 + 1024*2048 (fp32, 48 x 2048)
    const float* __restrict__ b1,
    const unsigned short* __restrict__ W2T, const float* __restrict__ b2,
    float* __restrict__ ceBuf) {
  const int t  = blockIdx.y;
  const int b0 = blockIdx.x * 16;
  const int tid = threadIdx.x;

  __shared__ float sL[16][DSTEP];
  __shared__ __align__(16) unsigned short hBf[16][HID];   // 64 KB
  __shared__ float part[8][3][16][16];                    // 24 KB
  __shared__ float logitsL[16][NPAD];

  for (int i = tid; i < 16 * DSTEP; i += 256)
    sL[i / DSTEP][i % DSTEP] =
        step48[((size_t)t * B_SZ + b0 + i / DSTEP) * DSTEP + (i % DSTEP)];
  __syncthreads();

  // h = Hbase + b1 + step48 @ W1[1024:]; swish; store bf16 to LDS
  #pragma unroll
  for (int j = 0; j < 8; ++j) {
    const int n = tid + j * 256;
    float acc[16];
    const float bb = b1[n];
    #pragma unroll
    for (int m = 0; m < 16; ++m) acc[m] = Hbase[(size_t)(b0 + m) * HID + n] + bb;
    for (int k = 0; k < DSTEP; ++k) {
      const float w = W1s[(size_t)k * HID + n];
      #pragma unroll
      for (int m = 0; m < 16; ++m) acc[m] = fmaf(sL[m][k], w, acc[m]);
    }
    #pragma unroll
    for (int m = 0; m < 16; ++m) {
      float h = acc[m];
      h = h / (1.f + __expf(-h));       // swish
      hBf[m][n] = f2bf(h);
    }
  }
  __syncthreads();

  // GEMM2: logits[16][48] = h(16x2048) @ W2(2048x48); K split over 8 waves
  const int lane = tid & 31, wave = tid >> 5;
  const int hl = lane >> 4, l16 = lane & 15;
  const int kBase = wave * 256;
  v8f zero = {};
  v8f acc2[3] = {zero, zero, zero};
  for (int c = 0; c < 8; ++c) {
    const int k0 = kBase + c * 32;
    Frag a;
    const unsigned short* pa = &hBf[l16][k0 + hl * 8];
    a.u[0] = *(const u32x4*)pa;
    a.u[1] = *(const u32x4*)(pa + 16);
    #pragma unroll
    for (int j = 0; j < 3; ++j) {
      Frag bw;
      const unsigned short* p = W2T + (size_t)(j * 16 + l16) * HID + k0 + hl * 8;
      bw.u[0] = *(const u32x4*)p;
      bw.u[1] = *(const u32x4*)(p + 16);
      acc2[j] = __builtin_amdgcn_wmma_f32_16x16x32_bf16(
          false, a.bf, false, bw.bf, (short)0, acc2[j], false, false);
    }
  }
  #pragma unroll
  for (int j = 0; j < 3; ++j)
    #pragma unroll
    for (int r = 0; r < 8; ++r)
      part[wave][j][r + hl * 8][l16] = acc2[j][r];
  __syncthreads();

  for (int idx = tid; idx < 768; idx += 256) {
    const int j = idx >> 8, rem = idx & 255;
    const int m = rem >> 4, nn = rem & 15;
    const int n = j * 16 + nn;
    float s = 0.f;
    #pragma unroll
    for (int w = 0; w < 8; ++w) s += part[w][j][m][nn];
    logitsL[m][n] = s + ((n < NACT) ? b2[n] : 0.f);
  }
  __syncthreads();

  if (tid < 16) {
    const int m = tid;
    float mx = -3.0e38f;
    for (int n = 0; n < NACT; ++n) mx = fmaxf(mx, logitsL[m][n]);
    float s = 0.f;
    for (int n = 0; n < NACT; ++n) s += __expf(logitsL[m][n] - mx);
    const float lse = mx + __logf(s);
    const int d = dacBuf[(size_t)t * B_SZ + b0 + m];
    ceBuf[(size_t)t * B_SZ + b0 + m] = lse - logitsL[m][d];
  }
}

// ---------------- final masked-mean reduction ----------------
__global__ void __launch_bounds__(1024) loss_kernel(const float* __restrict__ ceBuf,
                                                    const int* __restrict__ aliveBuf,
                                                    float* __restrict__ out) {
  __shared__ float red[1024];
  const int tid = threadIdx.x;
  float local = 0.f;
  for (int b = tid; b < B_SZ; b += 1024) {
    float s = 0.f, c = 0.f;
    for (int t = 0; t < T_STEPS; ++t) {
      const float a = aliveBuf[(size_t)t * B_SZ + b] ? 1.f : 0.f;
      s += ceBuf[(size_t)t * B_SZ + b] * a;
      c += a;
    }
    local += s / c;          // alive at t=0 always, c >= 1
  }
  red[tid] = local;
  __syncthreads();
  for (int off = 512; off > 0; off >>= 1) {
    if (tid < off) red[tid] += red[tid + off];
    __syncthreads();
  }
  if (tid == 0) out[0] = red[0] / (float)B_SZ;
}

// ---------------- launcher ----------------
extern "C" void kernel_launch(void* const* d_in, const int* in_sizes, int n_in,
                              void* d_out, int out_size, void* d_ws, size_t ws_size,
                              hipStream_t stream) {
  (void)in_sizes; (void)n_in; (void)out_size; (void)ws_size;
  const float* fc     = (const float*)d_in[0];
  const float* camera = (const float*)d_in[1];
  const int*   fb     = (const int*)d_in[2];
  const int*   lr     = (const int*)d_in[3];
  const int*   jp     = (const int*)d_in[4];
  const int*   ss     = (const int*)d_in[5];
  const int*   atk    = (const int*)d_in[6];
  const float* W1     = (const float*)d_in[7];
  const float* b1     = (const float*)d_in[8];
  const float* W2     = (const float*)d_in[9];
  const float* b2     = (const float*)d_in[10];
  float* out = (float*)d_out;

  char* ws = (char*)d_ws;
  size_t off = 0;
  unsigned short* fcB = (unsigned short*)(ws + off); off += (size_t)B_SZ * NI * 2;      // 8 MB
  unsigned short* W1T = (unsigned short*)(ws + off); off += (size_t)HID * NI * 2;       // 4 MB
  unsigned short* W2T = (unsigned short*)(ws + off); off += (size_t)NPAD * HID * 2;     // 192 KB
  float* step48  = (float*)(ws + off); off += (size_t)T_STEPS * B_SZ * DSTEP * 4;       // 15.7 MB
  int*   dacBuf  = (int*)(ws + off);   off += (size_t)T_STEPS * B_SZ * 4;
  int*   aliveBuf= (int*)(ws + off);   off += (size_t)T_STEPS * B_SZ * 4;
  float* Hbase   = (float*)(ws + off); off += (size_t)B_SZ * HID * 4;                   // 33.5 MB
  float* ceBuf   = (float*)(ws + off); off += (size_t)T_STEPS * B_SZ * 4;               // ~63 MB total

  cvt_fc_kernel <<<(B_SZ * NI + 255) / 256, 256, 0, stream>>>(fc, fcB);
  cvt_w1t_kernel<<<(HID * NI + 255) / 256, 256, 0, stream>>>(W1, W1T);
  cvt_w2t_kernel<<<(NPAD * HID + 255) / 256, 256, 0, stream>>>(W2, W2T);
  scan_kernel   <<<B_SZ / 256, 256, 0, stream>>>(camera, fb, lr, jp, ss, atk,
                                                 step48, dacBuf, aliveBuf);
  gemm_base_kernel<<<dim3(B_SZ / 128, HID / 128), 256, 0, stream>>>(fcB, W1T, Hbase);
  step_mlp_kernel <<<dim3(B_SZ / 16, T_STEPS), 256, 0, stream>>>(
      Hbase, step48, dacBuf, W1 + (size_t)NI * HID, b1, W2T, b2, ceBuf);
  loss_kernel<<<1, 1024, 0, stream>>>(ceBuf, aliveBuf, out);
}
